// AttentionDownsample_85907935855328
// MI455X (gfx1250) — matrix operand order repaired and checked by reading, not compiled, not run on hardware
//
#include <hip/hip_runtime.h>
#include <hip/hip_bf16.h>

typedef __bf16 v16bf __attribute__((ext_vector_type(16)));
typedef float  v8f   __attribute__((ext_vector_type(8)));

union V16 { v16bf v; __bf16 e[16]; };
union V8  { v8f   v; float  e[8];  };
union PK  { __bf16 h[2]; unsigned int u; };

#define WMMA_BF16(a, b, c) \
  __builtin_amdgcn_wmma_f32_16x16x32_bf16(false, (a), false, (b), (short)0, (c), false, false)

// 32B contiguous per-lane fragment load (B-fragment order / prob A rows)
__device__ __forceinline__ void load32B(V16& d, const __bf16* p) {
  *(ulonglong2*)&d.e[0] = *(const ulonglong2*)p;
  *(ulonglong2*)&d.e[8] = *(const ulonglong2*)(p + 8);
}
// A-fragment load from a row-major 32-wide span: K = e+hi*16
__device__ __forceinline__ void loadA(V16& d, const __bf16* row, int hi) {
  const __bf16* p = row + hi * 16;
  *(ulonglong2*)&d.e[0] = *(const ulonglong2*)p;
  *(ulonglong2*)&d.e[8] = *(const ulonglong2*)(p + 8);
}

// ---------------------------------------------------------------------------
// BN fold + swizzle weights into WMMA-B fragment order: [n/16][k/32][lane][e]
// ---------------------------------------------------------------------------
__global__ void fold_bn_swz(const float* __restrict__ W, const float* __restrict__ b0,
                            const float* __restrict__ g, const float* __restrict__ be,
                            const float* __restrict__ mu, const float* __restrict__ var,
                            int K, int N, __bf16* __restrict__ WF, float* __restrict__ bo) {
  int idx = blockIdx.x * 256 + threadIdx.x;
  if (idx < N) {
    float s = g[idx] * rsqrtf(var[idx] + 1e-3f);
    bo[idx] = (b0[idx] - mu[idx]) * s + be[idx];
  }
  int total = K * N;
  for (int i = idx; i < total; i += gridDim.x * 256) {
    int k = i / N, n = i % N;
    float s = g[n] * rsqrtf(var[n] + 1e-3f);
    size_t o = (((size_t)(n >> 4) * (K >> 5) + (k >> 5)) * 32 + (n & 15) + ((k >> 4 & 1) << 4))
                   * 16 + (k & 15);
    WF[o] = (__bf16)(W[i] * s);
  }
}

__global__ void cvt_bf16(const float* __restrict__ x, __bf16* __restrict__ xb, int n) {
  int i = blockIdx.x * 256 + threadIdx.x;
  if (i < n) xb[i] = (__bf16)x[i];
}

// ---------------------------------------------------------------------------
// kv GEMM: A via LDS, B fragments from swizzled global; scatters K into
// klayF[bh][sg][half][lane][e] (key j=s%32 -> half=j&1, n=j>>1, pad zeroed)
// and V into vbufF[bh][st][u][lane][e].
// ---------------------------------------------------------------------------
__global__ __launch_bounds__(256) void kv_gemm(const __bf16* __restrict__ xb,
                                               const __bf16* __restrict__ WF,
                                               const float* __restrict__ bias,
                                               __bf16* __restrict__ klayF,
                                               __bf16* __restrict__ vbufF) {
  __shared__ alignas(16) __bf16 As[128][32];
  const int tid = threadIdx.x, lane = tid & 31, wave = tid >> 5;
  const int ln = lane & 15, hi = lane >> 4;
  const int nbase = blockIdx.x * 64, mbase = blockIdx.y * 128;
  const int wm = wave >> 1, wn = wave & 1;
  V8 acc[2][2];
  for (int t = 0; t < 2; ++t)
    for (int u = 0; u < 2; ++u)
      for (int v = 0; v < 8; ++v) acc[t][u].e[v] = 0.f;

  for (int k0 = 0; k0 < 256; k0 += 32) {
    for (int t = 0; t < 4; ++t) {
      int idx = tid + t * 256, r = idx >> 3, cc = idx & 7;
      ((unsigned long long*)As)[idx] =
          *(const unsigned long long*)(xb + (size_t)(mbase + r) * 256 + k0 + cc * 4);
    }
    __syncthreads();
    V16 a[2], b[2];
    for (int t = 0; t < 2; ++t) loadA(a[t], &As[wm * 32 + t * 16 + ln][0], hi);
    for (int u = 0; u < 2; ++u) {
      size_t n16 = (nbase >> 4) + wn * 2 + u;
      load32B(b[u], WF + ((n16 * 8 + (k0 >> 5)) * 32 + lane) * 16);
    }
    for (int t = 0; t < 2; ++t)
      for (int u = 0; u < 2; ++u) acc[t][u].v = WMMA_BF16(a[t].v, b[u].v, acc[t][u].v);
    __syncthreads();
  }
  for (int t = 0; t < 2; ++t)
    for (int u = 0; u < 2; ++u)
      for (int v = 0; v < 8; ++v) {
        int m = mbase + wm * 32 + t * 16 + v + hi * 8;
        int c = nbase + wn * 32 + u * 16 + ln;
        float val = acc[t][u].e[v] + bias[c];
        int b_ = m >> 12, spos = m & 4095;
        int h = c / 80, ii = c % 80, bh = b_ * 8 + h;
        if (ii < 16) {
          int j = spos & 31, sg = spos >> 5;
          size_t base = ((((size_t)bh * 128 + sg) * 2 + (j & 1)) * 32) * 16;
          klayF[base + (size_t)(j >> 1) * 16 + ii] = (__bf16)val;
          klayF[base + (size_t)((j >> 1) + 16) * 16 + ii] = (__bf16)0.f;  // KD pad
        } else {
          int vd = ii - 16, st = spos >> 5, kk = spos & 31;
          vbufF[((((size_t)bh * 128 + st) * 4 + (vd >> 4)) * 32 + (vd & 15) + ((kk >> 4) << 4))
                    * 16 + (kk & 15)] = (__bf16)val;
        }
      }
}

// ---------------------------------------------------------------------------
// q GEMM (strided subsample, 0.25 scale folded, KD padded 16->32)
// ---------------------------------------------------------------------------
__global__ __launch_bounds__(256) void q_gemm(const __bf16* __restrict__ xb,
                                              const __bf16* __restrict__ WF,
                                              const float* __restrict__ bias,
                                              __bf16* __restrict__ qb) {
  __shared__ alignas(16) __bf16 As[128][32];
  const int tid = threadIdx.x, lane = tid & 31, wave = tid >> 5;
  const int ln = lane & 15, hi = lane >> 4;
  const int nbase = blockIdx.x * 64, mbase = blockIdx.y * 128;
  const int wm = wave >> 1, wn = wave & 1;
  V8 acc[2][2];
  for (int t = 0; t < 2; ++t)
    for (int u = 0; u < 2; ++u)
      for (int v = 0; v < 8; ++v) acc[t][u].e[v] = 0.f;

  for (int k0 = 0; k0 < 256; k0 += 32) {
    for (int t = 0; t < 4; ++t) {
      int idx = tid + t * 256, r = idx >> 3, cc = idx & 7;
      int m = mbase + r, b_ = m >> 10, q = m & 1023;
      int xrow = b_ * 4096 + ((q >> 5) << 7) + ((q & 31) << 1);
      ((unsigned long long*)As)[idx] =
          *(const unsigned long long*)(xb + (size_t)xrow * 256 + k0 + cc * 4);
    }
    __syncthreads();
    V16 a[2], b[2];
    for (int t = 0; t < 2; ++t) loadA(a[t], &As[wm * 32 + t * 16 + ln][0], hi);
    for (int u = 0; u < 2; ++u) {
      size_t n16 = (nbase >> 4) + wn * 2 + u;
      load32B(b[u], WF + ((n16 * 8 + (k0 >> 5)) * 32 + lane) * 16);
    }
    for (int t = 0; t < 2; ++t)
      for (int u = 0; u < 2; ++u) acc[t][u].v = WMMA_BF16(a[t].v, b[u].v, acc[t][u].v);
    __syncthreads();
  }
  for (int t = 0; t < 2; ++t)
    for (int u = 0; u < 2; ++u)
      for (int v = 0; v < 8; ++v) {
        int m = mbase + wm * 32 + t * 16 + v + hi * 8;
        int c = nbase + wn * 32 + u * 16 + ln;
        float val = (acc[t][u].e[v] + bias[c]) * 0.25f;
        int b_ = m >> 10, q = m & 1023;
        size_t base = ((size_t)(b_ * 8 + (c >> 4)) * 1024 + q) * 32;
        qb[base + (c & 15)] = (__bf16)val;
        qb[base + 16 + (c & 15)] = (__bf16)0.f;
      }
}

// ---------------------------------------------------------------------------
// Flash attention: one wave = 16-row q strip x SEQ=4096, 64 keys per online-
// softmax step (4 score + 8 AV WMMAs).  Denominator kept as per-lane partial
// sums (disjoint key ownership), butterfly-reduced once at the end.
// ---------------------------------------------------------------------------
__global__ __launch_bounds__(256) void attn_kernel(const __bf16* __restrict__ qb,
                                                   const __bf16* __restrict__ klayF,
                                                   const __bf16* __restrict__ vbufF,
                                                   __bf16* __restrict__ hsbuf) {
  __shared__ alignas(16) __bf16 pbuf[8][16][64];
  __shared__ alignas(16) __bf16 obuf[8][64][16];
  const int tid = threadIdx.x, wv = tid >> 5, lane = tid & 31;
  const int gw = blockIdx.x * 8 + wv;
  const int bh = gw >> 6, qtile = gw & 63;
  const int ln = lane & 15, hi = lane >> 4;

  V16 aq;
  loadA(aq, qb + ((size_t)bh * 1024 + qtile * 16 + ln) * 32, hi);
  V8 acc[4];
  for (int u = 0; u < 4; ++u)
    for (int v = 0; v < 8; ++v) acc[u].e[v] = 0.f;
  float mrow[8], lsum[8];
  for (int v = 0; v < 8; ++v) { mrow[v] = -3.0e30f; lsum[v] = 0.f; }
  V8 zz;
  for (int v = 0; v < 8; ++v) zz.e[v] = 0.f;

  for (int st = 0; st < 64; ++st) {  // 64 keys per step (two 32-key groups)
    V16 bk0, bk1, bk2, bk3;
    const __bf16* kp = klayF + (((size_t)bh * 128 + st * 2) << 10) + lane * 16;
    load32B(bk0, kp);          // group 2st,   keys 2n
    load32B(bk1, kp + 512);    // group 2st,   keys 2n+1
    load32B(bk2, kp + 1024);   // group 2st+1, keys 2n
    load32B(bk3, kp + 1536);   // group 2st+1, keys 2n+1
    V8 s0, s1, s2, s3;
    s0.v = WMMA_BF16(aq.v, bk0.v, zz.v);
    s1.v = WMMA_BF16(aq.v, bk1.v, zz.v);
    s2.v = WMMA_BF16(aq.v, bk2.v, zz.v);
    s3.v = WMMA_BF16(aq.v, bk3.v, zz.v);

    for (int v = 0; v < 8; ++v) {
      float a0 = s0.e[v], a1 = s1.e[v], a2 = s2.e[v], a3 = s3.e[v];
      float tm = fmaxf(fmaxf(a0, a1), fmaxf(a2, a3));
      for (int off = 8; off >= 1; off >>= 1) tm = fmaxf(tm, __shfl_xor(tm, off));
      float nm = fmaxf(mrow[v], tm);
      float corr = __expf(mrow[v] - nm);
      mrow[v] = nm;
      float p0 = __expf(a0 - nm), p1 = __expf(a1 - nm);
      float p2 = __expf(a2 - nm), p3 = __expf(a3 - nm);
      lsum[v] = lsum[v] * corr + (p0 + p1) + (p2 + p3);  // per-lane partial
      for (int u = 0; u < 4; ++u) acc[u].e[v] *= corr;
      int m = v + hi * 8;
      PK pk;
      pk.h[0] = (__bf16)p0; pk.h[1] = (__bf16)p1;
      *(unsigned int*)&pbuf[wv][m][2 * ln] = pk.u;
      pk.h[0] = (__bf16)p2; pk.h[1] = (__bf16)p3;
      *(unsigned int*)&pbuf[wv][m][32 + 2 * ln] = pk.u;
    }
    V16 pa0, pa1;
    loadA(pa0, &pbuf[wv][ln][0], hi);
    loadA(pa1, &pbuf[wv][ln][32], hi);
    const __bf16* vb = vbufF + (((size_t)bh * 128 + st * 2) << 11) + lane * 16;
    for (int u = 0; u < 4; ++u) {
      V16 bvf;
      load32B(bvf, vb + u * 512);
      acc[u].v = WMMA_BF16(pa0.v, bvf.v, acc[u].v);
    }
    for (int u = 0; u < 4; ++u) {
      V16 bvf;
      load32B(bvf, vb + 2048 + u * 512);
      acc[u].v = WMMA_BF16(pa1.v, bvf.v, acc[u].v);
    }
  }
  // one deferred butterfly reduction for the softmax denominator
  float lrow[8];
  for (int v = 0; v < 8; ++v) {
    float rs = lsum[v];
    for (int off = 8; off >= 1; off >>= 1) rs += __shfl_xor(rs, off);
    lrow[v] = rs;
  }
  // epilogue: 1/l, hard_swish, transpose via LDS, 32B stores in reshape layout
  const int b_ = bh >> 3, h = bh & 7;
  for (int u = 0; u < 4; ++u)
    for (int v = 0; v < 8; ++v) {
      float val = acc[u].e[v] / lrow[v];
      float hs = val * fminf(fmaxf(val + 3.f, 0.f), 6.f) * (1.f / 6.f);
      obuf[wv][u * 16 + ln][v + hi * 8] = (__bf16)hs;
    }
  for (int it = 0; it < 2; ++it) {
    int d = it * 32 + lane;
    ulonglong2 w0 = *(ulonglong2*)&obuf[wv][d][0];
    ulonglong2 w1 = *(ulonglong2*)&obuf[wv][d][8];
    int r = ((h * 64 + d) << 1) + (qtile >> 5);
    __bf16* dst = hsbuf + ((size_t)b_ * 1024 + r) * 512 + ((qtile * 16) & 511);
    *(ulonglong2*)dst = w0;
    *(ulonglong2*)(dst + 8) = w1;
  }
}

// ---------------------------------------------------------------------------
// proj GEMM: hsbuf[8192,512]bf16 @ WpF + bias -> fp32 out
// ---------------------------------------------------------------------------
__global__ __launch_bounds__(256) void proj_gemm(const __bf16* __restrict__ Ab,
                                                 const __bf16* __restrict__ WF,
                                                 const float* __restrict__ bias,
                                                 float* __restrict__ out) {
  __shared__ alignas(16) __bf16 As[128][32];
  const int tid = threadIdx.x, lane = tid & 31, wave = tid >> 5;
  const int ln = lane & 15, hi = lane >> 4;
  const int nbase = blockIdx.x * 64, mbase = blockIdx.y * 128;
  const int wm = wave >> 1, wn = wave & 1;
  V8 acc[2][2];
  for (int t = 0; t < 2; ++t)
    for (int u = 0; u < 2; ++u)
      for (int v = 0; v < 8; ++v) acc[t][u].e[v] = 0.f;

  for (int k0 = 0; k0 < 512; k0 += 32) {
    for (int t = 0; t < 4; ++t) {
      int idx = tid + t * 256, r = idx >> 3, cc = idx & 7;
      ((unsigned long long*)As)[idx] =
          *(const unsigned long long*)(Ab + (size_t)(mbase + r) * 512 + k0 + cc * 4);
    }
    __syncthreads();
    V16 a[2], b[2];
    for (int t = 0; t < 2; ++t) loadA(a[t], &As[wm * 32 + t * 16 + ln][0], hi);
    for (int u = 0; u < 2; ++u) {
      size_t n16 = (nbase >> 4) + wn * 2 + u;
      load32B(b[u], WF + ((n16 * 16 + (k0 >> 5)) * 32 + lane) * 16);
    }
    for (int t = 0; t < 2; ++t)
      for (int u = 0; u < 2; ++u) acc[t][u].v = WMMA_BF16(a[t].v, b[u].v, acc[t][u].v);
    __syncthreads();
  }
  for (int t = 0; t < 2; ++t)
    for (int u = 0; u < 2; ++u)
      for (int v = 0; v < 8; ++v) {
        int m = mbase + wm * 32 + t * 16 + v + hi * 8;
        int c = nbase + wn * 32 + u * 16 + ln;
        out[(size_t)m * 512 + c] = acc[t][u].e[v] + bias[c];
      }
}

// ---------------------------------------------------------------------------
extern "C" void kernel_launch(void* const* d_in, const int* in_sizes, int n_in,
                              void* d_out, int out_size, void* d_ws, size_t ws_size,
                              hipStream_t stream) {
  const float* x    = (const float*)d_in[0];
  const float* Wkv  = (const float*)d_in[1];
  const float* bkv  = (const float*)d_in[2];
  const float* g_kv = (const float*)d_in[3];
  const float* b_kv = (const float*)d_in[4];
  const float* m_kv = (const float*)d_in[5];
  const float* v_kv = (const float*)d_in[6];
  const float* Wq   = (const float*)d_in[7];
  const float* bq   = (const float*)d_in[8];
  const float* g_q  = (const float*)d_in[9];
  const float* b_q  = (const float*)d_in[10];
  const float* m_q  = (const float*)d_in[11];
  const float* v_q  = (const float*)d_in[12];
  const float* Wp   = (const float*)d_in[13];
  const float* bp   = (const float*)d_in[14];
  const float* g_p  = (const float*)d_in[15];
  const float* b_p  = (const float*)d_in[16];
  const float* m_p  = (const float*)d_in[17];
  const float* v_p  = (const float*)d_in[18];

  char* p = (char*)d_ws;
  auto carve = [&](size_t bytes) {
    char* r = p;
    p += (bytes + 255) & ~size_t(255);
    return r;
  };
  __bf16* WkvF = (__bf16*)carve((size_t)256 * 640 * 2);
  float*  bkv2 = (float*)carve(640 * 4);
  __bf16* WqF  = (__bf16*)carve((size_t)256 * 128 * 2);
  float*  bq2  = (float*)carve(128 * 4);
  __bf16* WpF  = (__bf16*)carve((size_t)512 * 512 * 2);
  float*  bp2  = (float*)carve(512 * 4);
  __bf16* xb   = (__bf16*)carve((size_t)32768 * 256 * 2);
  __bf16* klayF= (__bf16*)carve((size_t)64 * 128 * 2 * 32 * 16 * 2);   // 16 MB
  __bf16* vbufF= (__bf16*)carve((size_t)64 * 128 * 4 * 32 * 16 * 2);   // 32 MB
  __bf16* qbuf = (__bf16*)carve((size_t)64 * 1024 * 32 * 2);
  __bf16* hsb  = (__bf16*)carve((size_t)8 * 1024 * 512 * 2);

  fold_bn_swz<<<(256 * 640 + 255) / 256, 256, 0, stream>>>(Wkv, bkv, g_kv, b_kv, m_kv, v_kv,
                                                           256, 640, WkvF, bkv2);
  fold_bn_swz<<<(256 * 128 + 255) / 256, 256, 0, stream>>>(Wq, bq, g_q, b_q, m_q, v_q,
                                                           256, 128, WqF, bq2);
  fold_bn_swz<<<(512 * 512 + 255) / 256, 256, 0, stream>>>(Wp, bp, g_p, b_p, m_p, v_p,
                                                           512, 512, WpF, bp2);
  cvt_bf16<<<(8 * 4096 * 256) / 256, 256, 0, stream>>>(x, xb, 8 * 4096 * 256);

  kv_gemm<<<dim3(10, 256), 256, 0, stream>>>(xb, WkvF, bkv2, klayF, vbufF);
  q_gemm<<<dim3(2, 64), 256, 0, stream>>>(xb, WqF, bq2, qbuf);
  attn_kernel<<<512, 256, 0, stream>>>(qbuf, klayF, vbufF, hsb);
  proj_gemm<<<dim3(8, 64), 256, 0, stream>>>(hsb, WpF, bp2, (float*)d_out);
}
